// Attention_15135464751770
// MI455X (gfx1250) — matrix operand order repaired and checked by reading, not compiled, not run on hardware
//
#include <hip/hip_runtime.h>
#include <math.h>

// ---------------- problem constants ----------------
#define BATCH 128
#define NTOK  196
#define DIMX  384
#define NHEAD 8
#define KD    32
#define DV    128
#define HOUT  1536     // 8*(32+32+128)
#define DH    1024     // 8*128
#define TOT   (BATCH*NTOK)   // 25088
#define NPAD  224            // 7 K-chunks of 32 for the AV GEMM
#define SCALE_QK 0.17677669529663687f  // 32^-0.5
#define LOG2E 1.4426950408889634f

typedef __bf16 bf16;
typedef __attribute__((ext_vector_type(8)))  bf16  bf16x8;
typedef __attribute__((ext_vector_type(16))) bf16  bf16x16;
typedef __attribute__((ext_vector_type(8)))  float f32x8;

// ---------------- WMMA fragment helpers (contiguous k = 16*half + s in both A and B) ----
static __device__ __forceinline__ bf16x16 cat8(bf16x8 lo, bf16x8 hi) {
  return __builtin_shufflevector(lo, hi, 0,1,2,3,4,5,6,7,8,9,10,11,12,13,14,15);
}

// A-matrix 16x32 bf16: lane%16 = row M; elems 0..15 = K[half*16 .. +15]
// (contiguous, two adjacent 16B loads)
static __device__ __forceinline__ bf16x16
load_a_frag(const bf16* __restrict__ base, int ldk, int m, int k0, int lane) {
  int half = lane >> 4;
  const bf16* p = base + (long)m * ldk + k0;
  bf16x8 lo = *reinterpret_cast<const bf16x8*>(p + half * 16);
  bf16x8 hi = *reinterpret_cast<const bf16x8*>(p + half * 16 + 8);
  return cat8(lo, hi);
}

// B-matrix 32x16 bf16: lane%16 = col N; per-lane K = half*16 .. half*16+15
// (contiguous, two adjacent 16B loads).  "n" row of the K-major (transposed) array.
static __device__ __forceinline__ bf16x16
load_b_frag(const bf16* __restrict__ base, int ldk, int n, int k0, int lane) {
  int half = lane >> 4;
  const bf16* p = base + (long)n * ldk + k0 + half * 16;
  bf16x8 lo = *reinterpret_cast<const bf16x8*>(p);
  bf16x8 hi = *reinterpret_cast<const bf16x8*>(p + 8);
  return cat8(lo, hi);
}

static __device__ __forceinline__ f32x8 wmma_bf16(bf16x16 a, bf16x16 b, f32x8 c) {
  return __builtin_amdgcn_wmma_f32_16x16x32_bf16(false, a, false, b, (short)0, c,
                                                 false, false);
}

// ---------------- kernel 1: weight cast + bias gather + V-pad zero ------------
__global__ void k_prep(const float* __restrict__ qkv_w,
                       const float* __restrict__ proj_w,
                       const float* __restrict__ attn_biases,
                       const int*   __restrict__ bias_idxs, int n_off,
                       bf16* __restrict__ qkvw_bf, bf16* __restrict__ projw_bf,
                       float* __restrict__ bias_mat, bf16* __restrict__ vbuf_t) {
  const long n1 = (long)HOUT * DIMX;          // 589824
  const long n2 = (long)DIMX * DH;            // 393216
  const long n3 = (long)NHEAD * NTOK * NTOK;  // 307328
  const long n4 = (long)BATCH * NHEAD * DV * (NPAD - NTOK);
  const long ntot = n1 + n2 + n3 + n4;
  long stride = (long)gridDim.x * blockDim.x;
  for (long t = (long)blockIdx.x * blockDim.x + threadIdx.x; t < ntot; t += stride) {
    if (t < n1) {
      qkvw_bf[t] = (bf16)qkv_w[t];
    } else if (t < n1 + n2) {
      long u = t - n1;
      projw_bf[u] = (bf16)proj_w[u];
    } else if (t < n1 + n2 + n3) {
      long u = t - n1 - n2;
      int h = (int)(u / (NTOK * NTOK));
      int ij = (int)(u % (NTOK * NTOK));
      bias_mat[u] = attn_biases[(long)h * n_off + bias_idxs[ij]];
    } else {
      long u = t - n1 - n2 - n3;
      long row = u / (NPAD - NTOK);
      int  j   = NTOK + (int)(u % (NPAD - NTOK));
      vbuf_t[row * NPAD + j] = (bf16)0.0f;
    }
  }
}

// ---------------- kernel 2: LayerNorm -> bf16 (one wave32 per token) ----------
__global__ void __launch_bounds__(256) k_ln(const float* __restrict__ x,
                                            const float* __restrict__ ln_w,
                                            const float* __restrict__ ln_b,
                                            bf16* __restrict__ xn) {
  int wave = threadIdx.x >> 5;
  int lane = threadIdx.x & 31;
  long t = (long)blockIdx.x * 8 + wave;       // 3136 blocks * 8 = 25088 exactly
  const float* xp = x + t * DIMX;
  float v[12];
  float s = 0.f;
#pragma unroll
  for (int i = 0; i < 12; i++) { v[i] = xp[lane + 32 * i]; s += v[i]; }
#pragma unroll
  for (int m = 16; m >= 1; m >>= 1) s += __shfl_xor(s, m, 32);
  float mu = s * (1.0f / DIMX);
  float q = 0.f;
#pragma unroll
  for (int i = 0; i < 12; i++) { float d = v[i] - mu; q += d * d; }
#pragma unroll
  for (int m = 16; m >= 1; m >>= 1) q += __shfl_xor(q, m, 32);
  float rstd = rsqrtf(q * (1.0f / DIMX) + 1e-5f);
  bf16* op = xn + t * DIMX;
#pragma unroll
  for (int i = 0; i < 12; i++) {
    int c = lane + 32 * i;
    op[c] = (bf16)((v[i] - mu) * rstd * ln_w[c] + ln_b[c]);
  }
}

// ---------------- kernel 3: QKV GEMM (25088x384 @ 384x1536) -------------------
// per-wave 64x32 tile, 4x2 WMMA microtiles (8 WMMA / 6 fragment loads per K step
// -> ~21 FLOP/B from L2).  K loop of 12.
// epilogue scatters into q[B,H,N,32], k[B,H,N,32], vT[B,H,128,224] (bf16).
__global__ void __launch_bounds__(256) k_qkv(const bf16* __restrict__ xn,
                                             const bf16* __restrict__ wq,
                                             const float* __restrict__ qkv_b,
                                             bf16* __restrict__ qbuf,
                                             bf16* __restrict__ kbuf,
                                             bf16* __restrict__ vbuf_t) {
  int lane = threadIdx.x & 31;
  int wv   = threadIdx.x >> 5;
  int tile = blockIdx.x * 8 + wv;             // 2352*8 = 18816 = 392*48 tiles
  int nt = tile % 48, mt = tile / 48;
  int m0 = mt * 64, n0 = nt * 32;
  int l16 = lane & 15, half = lane >> 4;
  f32x8 acc[4][2] = {};
  for (int kt = 0; kt < 12; kt++) {
    int k0 = kt * 32;
    bf16x16 b0 = load_b_frag(wq, DIMX, n0 + l16,      k0, lane);
    bf16x16 b1 = load_b_frag(wq, DIMX, n0 + 16 + l16, k0, lane);
#pragma unroll
    for (int i = 0; i < 4; i++) {
      bf16x16 a = load_a_frag(xn, DIMX, m0 + 16 * i + l16, k0, lane);
      acc[i][0] = wmma_bf16(a, b0, acc[i][0]);
      acc[i][1] = wmma_bf16(a, b1, acc[i][1]);
    }
  }
#pragma unroll
  for (int i = 0; i < 4; i++)
#pragma unroll
    for (int j = 0; j < 2; j++) {
      int c = n0 + j * 16 + l16;              // output feature
      float bias = qkv_b[c];
      int h = c / 192, rr = c % 192;          // per-head: [q32 | k32 | v128]
#pragma unroll
      for (int r = 0; r < 8; r++) {
        int m = m0 + i * 16 + r + 8 * half;   // token row (C/D layout)
        int b = m / NTOK, n = m % NTOK;
        bf16 bv = (bf16)(acc[i][j][r] + bias);
        if (rr < KD)
          qbuf[(((long)b * NHEAD + h) * NTOK + n) * KD + rr] = bv;
        else if (rr < 2 * KD)
          kbuf[(((long)b * NHEAD + h) * NTOK + n) * KD + (rr - KD)] = bv;
        else
          vbuf_t[(((long)b * NHEAD + h) * DV + (rr - 2 * KD)) * NPAD + n] = bv;
      }
    }
}

// ---------------- kernel 4: fused attention per (b,h) -------------------------
// 13 waves; wave w owns query rows [16w,16w+16). S held in 13 f32x8 accumulators,
// half-wave shuffle softmax, P relaid via per-wave LDS tile into A-fragments.
__global__ void __launch_bounds__(416) k_attn(const bf16* __restrict__ qbuf,
                                              const bf16* __restrict__ kbuf,
                                              const bf16* __restrict__ vbuf_t,
                                              const float* __restrict__ bias_mat,
                                              bf16* __restrict__ attnout) {
  __shared__ __align__(16) bf16 p_lds[13][16][32];
  int bh = blockIdx.x;                        // b*8 + h
  int h = bh & 7, b = bh >> 3;
  int lane = threadIdx.x & 31;
  int w    = threadIdx.x >> 5;                // row tile 0..12
  int l16 = lane & 15, half = lane >> 4;
  const bf16* qb = qbuf + (long)bh * NTOK * KD;
  const bf16* kb = kbuf + (long)bh * NTOK * KD;
  const bf16* vb = vbuf_t + (long)bh * DV * NPAD;
  const float* bm = bias_mat + (long)h * NTOK * NTOK;

  int mrow = w * 16 + l16; if (mrow > NTOK - 1) mrow = NTOK - 1;
  bf16x16 aq = load_a_frag(qb, KD, mrow, 0, lane);   // KEY_DIM==32: one fragment

  f32x8 s[13];
#pragma unroll
  for (int ct = 0; ct < 13; ct++) {                  // S = q k^T, one WMMA/tile
    int j = ct * 16 + l16; int jc = j < NTOK ? j : NTOK - 1;
    bf16x16 bk = load_b_frag(kb, KD, jc, 0, lane);
    f32x8 z = {};
    s[ct] = wmma_bf16(aq, bk, z);
  }
  // logits: scale + relative-position bias, -inf on padded keys
#pragma unroll
  for (int ct = 0; ct < 13; ct++) {
    int j = ct * 16 + l16;
#pragma unroll
    for (int r = 0; r < 8; r++) {
      int i = w * 16 + r + 8 * half;
      int ic = i < NTOK ? i : NTOK - 1;
      if (j < NTOK)
        s[ct][r] = s[ct][r] * SCALE_QK + bm[(long)ic * NTOK + j];
      else
        s[ct][r] = -1e30f;
    }
  }
  // softmax across keys: reduce over 13 frags + 16 lanes of this half-wave
  float inv[8];
#pragma unroll
  for (int r = 0; r < 8; r++) {
    float mx = s[0][r];
#pragma unroll
    for (int ct = 1; ct < 13; ct++) mx = fmaxf(mx, s[ct][r]);
#pragma unroll
    for (int m = 1; m < 16; m <<= 1) mx = fmaxf(mx, __shfl_xor(mx, m, 32));
    float sum = 0.f;
#pragma unroll
    for (int ct = 0; ct < 13; ct++) {
      float e = exp2f((s[ct][r] - mx) * LOG2E);
      s[ct][r] = e; sum += e;
    }
#pragma unroll
    for (int m = 1; m < 16; m <<= 1) sum += __shfl_xor(sum, m, 32);
    inv[r] = 1.0f / sum;
  }
  // O = P @ V : 7 K-chunks of 32 keys, 8 d-tiles
  f32x8 o[8] = {};
  for (int kk = 0; kk < 7; kk++) {
#pragma unroll
    for (int tt = 0; tt < 2; tt++) {
      int ct = kk * 2 + tt;
#pragma unroll
      for (int r = 0; r < 8; r++) {
        bf16 pv = (ct < 13) ? (bf16)(s[ct][r] * inv[r]) : (bf16)0.0f;
        p_lds[w][r + 8 * half][tt * 16 + l16] = pv;   // D-layout -> row-major
      }
    }
    __syncthreads();
    bf16x16 pa = load_a_frag(&p_lds[w][0][0], 32, l16, 0, lane);
#pragma unroll
    for (int dt = 0; dt < 8; dt++) {
      bf16x16 bv = load_b_frag(vb, NPAD, dt * 16 + l16, kk * 32, lane);
      o[dt] = wmma_bf16(pa, bv, o[dt]);
    }
    __syncthreads();
  }
  // store to attnout [B, N, H*128] bf16 (token-major for the proj GEMM)
#pragma unroll
  for (int dt = 0; dt < 8; dt++) {
    int d = dt * 16 + l16;
#pragma unroll
    for (int r = 0; r < 8; r++) {
      int i = w * 16 + r + 8 * half;
      if (i < NTOK)
        attnout[((long)b * NTOK + i) * DH + h * DV + d] = (bf16)o[dt][r];
    }
  }
}

// ---------------- kernel 5: proj GEMM (25088x1024 @ 1024x384) + bias ----------
// per-wave 64x32 tile, 4x2 WMMA microtiles, K loop of 32.
__global__ void __launch_bounds__(256) k_proj(const bf16* __restrict__ ao,
                                              const bf16* __restrict__ wp,
                                              const float* __restrict__ proj_b,
                                              float* __restrict__ out) {
  int lane = threadIdx.x & 31;
  int wv   = threadIdx.x >> 5;
  int tile = blockIdx.x * 8 + wv;             // 588*8 = 4704 = 392*12 tiles
  int nt = tile % 12, mt = tile / 12;
  int m0 = mt * 64, n0 = nt * 32;
  int l16 = lane & 15, half = lane >> 4;
  f32x8 acc[4][2] = {};
  for (int kt = 0; kt < 32; kt++) {
    int k0 = kt * 32;
    bf16x16 b0 = load_b_frag(wp, DH, n0 + l16,      k0, lane);
    bf16x16 b1 = load_b_frag(wp, DH, n0 + 16 + l16, k0, lane);
#pragma unroll
    for (int i = 0; i < 4; i++) {
      bf16x16 a = load_a_frag(ao, DH, m0 + 16 * i + l16, k0, lane);
      acc[i][0] = wmma_bf16(a, b0, acc[i][0]);
      acc[i][1] = wmma_bf16(a, b1, acc[i][1]);
    }
  }
#pragma unroll
  for (int i = 0; i < 4; i++)
#pragma unroll
    for (int j = 0; j < 2; j++) {
      int c = n0 + j * 16 + l16;
      float pb = proj_b[c];
#pragma unroll
      for (int r = 0; r < 8; r++) {
        int m = m0 + i * 16 + r + 8 * half;
        out[(long)m * DIMX + c] = acc[i][j][r] + pb;
      }
    }
}

// ---------------- launcher ----------------------------------------------------
extern "C" void kernel_launch(void* const* d_in, const int* in_sizes, int n_in,
                              void* d_out, int out_size, void* d_ws, size_t ws_size,
                              hipStream_t stream) {
  const float* x      = (const float*)d_in[0];
  const float* ln_w   = (const float*)d_in[1];
  const float* ln_b   = (const float*)d_in[2];
  const float* qkv_w  = (const float*)d_in[3];
  const float* qkv_b  = (const float*)d_in[4];
  const float* proj_w = (const float*)d_in[5];
  const float* proj_b = (const float*)d_in[6];
  const float* abias  = (const float*)d_in[7];
  const int*   bidx   = (const int*)d_in[8];
  float* out = (float*)d_out;
  int n_off = in_sizes[7] / NHEAD;

  char* ws = (char*)d_ws;
  // region A: xn_bf + qkvw_bf live only until k_qkv; attnout (written later)
  // aliases over them.  All offsets are 256B-aligned.
  const size_t off_xn     = 0;                               // 19,267,584 B
  const size_t off_qkvw   = (size_t)TOT * DIMX * 2;          // +1,179,648 B
  const size_t off_attno  = 0;                               // 51,380,224 B (aliases A)
  const size_t regA       = (size_t)TOT * DH * 2;
  const size_t off_projw  = regA;                            //   786,432 B
  const size_t off_biasm  = off_projw + (size_t)DIMX * DH * 2;           // 1,229,312 B
  const size_t off_qbuf   = off_biasm + (size_t)NHEAD * NTOK * NTOK * 4; // 12,845,056 B
  const size_t off_kbuf   = off_qbuf + (size_t)BATCH * NHEAD * NTOK * KD * 2;
  const size_t off_vbuf   = off_kbuf + (size_t)BATCH * NHEAD * NTOK * KD * 2;

  bf16*  xn_bf   = (bf16*)(ws + off_xn);
  bf16*  qkvw_bf = (bf16*)(ws + off_qkvw);
  bf16*  projw_bf= (bf16*)(ws + off_projw);
  float* bias_mat= (float*)(ws + off_biasm);
  bf16*  qbuf    = (bf16*)(ws + off_qbuf);
  bf16*  kbuf    = (bf16*)(ws + off_kbuf);
  bf16*  vbuf_t  = (bf16*)(ws + off_vbuf);
  bf16*  attnout = (bf16*)(ws + off_attno);

  k_prep<<<4096, 256, 0, stream>>>(qkv_w, proj_w, abias, bidx, n_off,
                                   qkvw_bf, projw_bf, bias_mat, vbuf_t);
  k_ln<<<TOT / 8, 256, 0, stream>>>(x, ln_w, ln_b, xn_bf);
  k_qkv<<<(392 * 48) / 8, 256, 0, stream>>>(xn_bf, qkvw_bf, qkv_b,
                                            qbuf, kbuf, vbuf_t);
  k_attn<<<BATCH * NHEAD, 13 * 32, 0, stream>>>(qbuf, kbuf, vbuf_t,
                                                bias_mat, attnout);
  k_proj<<<(392 * 12) / 8, 256, 0, stream>>>(attnout, projw_bf, proj_b, out);
}
